// LstmTrainStep_29858612642440
// MI455X (gfx1250) — matrix operand order repaired and checked by reading, not compiled, not run on hardware
//
#include <hip/hip_runtime.h>
#include <hip/hip_bf16.h>

// Problem dims (fixed by reference)
#define B_   2048
#define IH   512        // I == H == 512
#define NG   2048       // 4*H
#define TT   64

// GEMM tiling
#define TM   128
#define TN   128
#define LDA  48         // LDS row stride in ushorts (96B = 6*16B, keeps b128 aligned)

typedef __attribute__((ext_vector_type(16))) __bf16 v16bf;
typedef __attribute__((ext_vector_type(8)))  float  v8f;

__device__ __forceinline__ unsigned short f2bf(float f) {
    unsigned int u = __builtin_bit_cast(unsigned int, f);
    unsigned int r = u + 0x7FFFu + ((u >> 16) & 1u);   // round-to-nearest-even
    return (unsigned short)(r >> 16);
}

__device__ __forceinline__ unsigned ldsoff(const void* p) {
    return (unsigned)(unsigned long long)p;            // low 32 bits = LDS offset
}

// CDNA5 async global->LDS copy, 16 bytes per lane, tracked by ASYNCcnt
__device__ __forceinline__ void async_b128(const unsigned short* g, unsigned lds) {
    asm volatile("global_load_async_to_lds_b128 %0, %1, off"
                 :: "v"(lds), "v"(g) : "memory");
}

__device__ __forceinline__ void wait_async0() {
#if __has_builtin(__builtin_amdgcn_s_wait_asynccnt)
    __builtin_amdgcn_s_wait_asynccnt(0);
#else
    asm volatile("s_wait_asynccnt 0" ::: "memory");
#endif
}

// ---------------------------------------------------------------------------
// fp32 -> bf16 bulk convert, 4 elems/thread (latent and weights; layout kept)
// ---------------------------------------------------------------------------
__global__ __launch_bounds__(256)
void cvt_bf16_x4(const float* __restrict__ src, unsigned short* __restrict__ dst) {
    size_t i = ((size_t)blockIdx.x * 256 + threadIdx.x) * 4;
    float4 v = *(const float4*)(src + i);
    uint2 p;
    p.x = (unsigned)f2bf(v.x) | ((unsigned)f2bf(v.y) << 16);
    p.y = (unsigned)f2bf(v.z) | ((unsigned)f2bf(v.w) << 16);
    *(uint2*)(dst + i) = p;
}

// ---------------------------------------------------------------------------
// h0 -> bf16 with reset_mask[:, 0] applied
// ---------------------------------------------------------------------------
__global__ __launch_bounds__(256)
void h0_prep(const float* __restrict__ h0, const unsigned char* __restrict__ mask,
             unsigned short* __restrict__ hbf) {
    int idx = blockIdx.x * 256 + threadIdx.x;   // 0 .. B*H-1
    int b = idx >> 9;
    hbf[idx] = mask[(size_t)b * TT] ? (unsigned short)0 : f2bf(h0[idx]);
}

// ---------------------------------------------------------------------------
// Per-step gates GEMM:  gates[B,NG] = Xbf[B,IH] @ WihB^T + Hbf[B,IH] @ WhhB^T
// All operands bf16; A sources [row][k], B sources (weights) native [n][k].
// 256 threads = 8 waves; wave grid 4(M)x2(N); each wave owns 2x4 16x16 tiles.
// Double-buffered LDS staging via async global->LDS b128 copies; running
// source pointers (stride 32 ushorts per K-step, one swap at the X->H seam).
// ---------------------------------------------------------------------------
__global__ __launch_bounds__(256)
void lstm_gates_gemm(const unsigned short* __restrict__ Xbf,   // [B][IH] bf16
                     const unsigned short* __restrict__ Hbf,   // [B][IH] bf16 (masked)
                     const unsigned short* __restrict__ WihB,  // [NG][IH] bf16
                     const unsigned short* __restrict__ WhhB,  // [NG][IH] bf16
                     float* __restrict__ gates) {              // [B][NG]
    __shared__ unsigned short Asl[2][TM][LDA];   // [buf][m][k]
    __shared__ unsigned short Bsl[2][TN][LDA];   // [buf][n][k]

    const int tid  = threadIdx.x;
    const int lane = tid & 31;
    const int wave = tid >> 5;      // 0..7
    const int wm   = wave & 3;      // 32 rows per wave slot
    const int wn   = wave >> 2;     // 64 cols per wave slot

    const int m0 = blockIdx.y * TM;
    const int n0 = blockIdx.x * TN;

    const int lg = lane >> 4;       // lane group (0/1)
    const int lm = lane & 15;

    v8f acc[2][4];
#pragma unroll
    for (int i = 0; i < 2; ++i)
#pragma unroll
        for (int j = 0; j < 4; ++j)
            acc[i][j] = (v8f){0.f,0.f,0.f,0.f,0.f,0.f,0.f,0.f};

    // staging: 4 lanes/row * 16B, 64 rows/pass, 2 passes -> 4 asyncs/thread/tile-set
    const int srow = tid >> 2;              // 0..63
    const int scol = (tid & 3) * 8;         // k offset in ushorts

    // per-thread LDS destinations (constant across iterations, per buffer)
    const unsigned ldsA0 = ldsoff(&Asl[0][srow][scol]);
    const unsigned ldsA1 = ldsoff(&Asl[0][srow + 64][scol]);
    const unsigned ldsB0 = ldsoff(&Bsl[0][srow][scol]);
    const unsigned ldsB1 = ldsoff(&Bsl[0][srow + 64][scol]);
    const unsigned bufSz = (unsigned)sizeof(Asl[0]);   // 12288 B (same for Bsl)

    // running per-thread global source pointers (advance 32 ushorts per K-step)
    const unsigned short* aSrc = Xbf  + (size_t)(m0 + srow) * IH + scol;
    const unsigned short* bSrc = WihB + (size_t)(n0 + srow) * IH + scol;

    auto stageP = [&](const unsigned short* a, const unsigned short* b, int buf) {
        const unsigned o = (unsigned)buf * bufSz;
        async_b128(a,            ldsA0 + o);
        async_b128(a + 64 * IH,  ldsA1 + o);
        async_b128(b,            ldsB0 + o);
        async_b128(b + 64 * IH,  ldsB1 + o);
    };

    stageP(aSrc, bSrc, 0);

#pragma unroll 2
    for (int kb = 0; kb < 32; ++kb) {
        const int cur = kb & 1;
        wait_async0();          // own wave's copies into buf[cur] landed
        __syncthreads();        // everyone's landed; everyone done reading buf[cur^1]
        if (kb < 31) {          // overlap next staging with compute
            if (kb + 1 == 16) { // X->H seam: swap sources, reset K position
                aSrc = Hbf  + (size_t)(m0 + srow) * IH + scol;
                bSrc = WhhB + (size_t)(n0 + srow) * IH + scol;
            } else {
                aSrc += 32;
                bSrc += 32;
            }
            stageP(aSrc, bSrc, cur ^ 1);
        }

        // Load ALL fragments first so the 12 ds_load_b128 pipeline, then WMMA.
        // A frag (16x32 bf16): lane m=lm; elems 0-7 -> K=lg*8+e, 8-15 -> 16+lg*8+e
        v16bf afrag[2];
#pragma unroll
        for (int mi = 0; mi < 2; ++mi) {
            int r = wm * 32 + mi * 16 + lm;
            union { v16bf v; uint4 q[2]; } u;
            u.q[0] = *(const uint4*)&Asl[cur][r][lg * 8];
            u.q[1] = *(const uint4*)&Asl[cur][r][16 + lg * 8];
            afrag[mi] = u.v;
        }
        // B frag (32x16 bf16): lane n=lm; elems 0-7 -> K=lg*8+e, 8-15 -> 16+lg*8+e
        v16bf bfrag[4];
#pragma unroll
        for (int ni = 0; ni < 4; ++ni) {
            int c = wn * 64 + ni * 16 + lm;
            union { v16bf v; uint4 q[2]; } u;
            u.q[0] = *(const uint4*)&Bsl[cur][c][lg * 8];
            u.q[1] = *(const uint4*)&Bsl[cur][c][16 + lg * 8];
            bfrag[ni] = u.v;
        }
#pragma unroll
        for (int ni = 0; ni < 4; ++ni)
#pragma unroll
            for (int mi = 0; mi < 2; ++mi)
                acc[mi][ni] = __builtin_amdgcn_wmma_f32_16x16x32_bf16(
                    false, afrag[mi], false, bfrag[ni], (short)0, acc[mi][ni],
                    false, false);
    }

    // store C: VGPR v -> row = base + v + 8*lg, col = base + lm
#pragma unroll
    for (int mi = 0; mi < 2; ++mi)
#pragma unroll
        for (int ni = 0; ni < 4; ++ni) {
            int row = m0 + wm * 32 + mi * 16 + 8 * lg;
            int col = n0 + wn * 64 + ni * 16 + lm;
            float* dst = gates + (size_t)row * NG + col;
#pragma unroll
            for (int v = 0; v < 8; ++v)
                dst[(size_t)v * NG] = acc[mi][ni][v];
        }
}

// ---------------------------------------------------------------------------
// Elementwise LSTM cell; also emits bf16 h pre-masked for the NEXT step
// ---------------------------------------------------------------------------
__global__ __launch_bounds__(256)
void lstm_cell(const float* __restrict__ gates, const float* __restrict__ c0,
               const unsigned char* __restrict__ mask,
               const float* __restrict__ b_ih, const float* __restrict__ b_hh,
               float* __restrict__ cbuf, float* __restrict__ hout,
               unsigned short* __restrict__ hbf,
               int t, int firstStep, int notLast) {
    int idx = blockIdx.x * 256 + threadIdx.x;    // 0 .. B*H-1
    int b = idx >> 9;
    int h = idx & (IH - 1);
    float c = firstStep ? c0[idx] : cbuf[idx];
    if (mask[(size_t)b * TT + t]) c = 0.f;
    const float* g = gates + (size_t)b * NG;
    float gi = g[h]          + b_ih[h]          + b_hh[h];
    float gf = g[h + IH]     + b_ih[h + IH]     + b_hh[h + IH];
    float gg = g[h + 2*IH]   + b_ih[h + 2*IH]   + b_hh[h + 2*IH];
    float go = g[h + 3*IH]   + b_ih[h + 3*IH]   + b_hh[h + 3*IH];
    float si = 1.f / (1.f + __expf(-gi));
    float sf = 1.f / (1.f + __expf(-gf));
    float so = 1.f / (1.f + __expf(-go));
    c = sf * c + si * tanhf(gg);
    float hn = so * tanhf(c);
    cbuf[idx] = c;
    hout[idx] = hn;
    if (notLast) {
        unsigned short hb = f2bf(hn);
        if (mask[(size_t)b * TT + t + 1]) hb = 0;   // pre-apply next step's reset
        hbf[idx] = hb;
    }
}

// ---------------------------------------------------------------------------
extern "C" void kernel_launch(void* const* d_in, const int* in_sizes, int n_in,
                              void* d_out, int out_size, void* d_ws, size_t ws_size,
                              hipStream_t stream) {
    (void)in_sizes; (void)n_in; (void)out_size; (void)ws_size;
    const float*         latent = (const float*)d_in[0];          // [T,B,I]
    const float*         h0     = (const float*)d_in[1];          // [1,B,H]
    const float*         c0     = (const float*)d_in[2];          // [1,B,H]
    const unsigned char* mask   = (const unsigned char*)d_in[3];  // [1,B,T,1] bool
    const float*         W_ih   = (const float*)d_in[4];          // [4H,I]
    const float*         W_hh   = (const float*)d_in[5];          // [4H,H]
    const float*         b_ih   = (const float*)d_in[6];          // [4H]
    const float*         b_hh   = (const float*)d_in[7];          // [4H]

    float* hidden = (float*)d_out;                                // [T,B,H]
    float* h_n    = hidden + (size_t)TT * B_ * IH;                // [B,H]
    float* c_n    = h_n + (size_t)B_ * IH;                        // [B,H]

    // workspace: Xbf(128MB) | Hbf(2MB) | WihB(2MB) | WhhB(2MB) | gates(16MB) | cbuf(4MB)
    unsigned short* Xbf  = (unsigned short*)d_ws;
    unsigned short* Hbf  = Xbf  + (size_t)TT * B_ * IH;
    unsigned short* WihB = Hbf  + (size_t)B_ * IH;
    unsigned short* WhhB = WihB + (size_t)NG * IH;
    float* gates = (float*)(WhhB + (size_t)NG * IH);
    float* cbuf  = gates + (size_t)B_ * NG;

    // one-time conversions (off the serial path; weights then live in L2)
    cvt_bf16_x4<<<((size_t)TT * B_ * IH) / 1024, 256, 0, stream>>>(latent, Xbf);
    cvt_bf16_x4<<<((size_t)NG * IH) / 1024, 256, 0, stream>>>(W_ih, WihB);
    cvt_bf16_x4<<<((size_t)NG * IH) / 1024, 256, 0, stream>>>(W_hh, WhhB);
    h0_prep<<<(B_ * IH) / 256, 256, 0, stream>>>(h0, mask, Hbf);

    dim3 gemmGrid(NG / TN, B_ / TM);   // 16 x 16
    for (int t = 0; t < TT; ++t) {
        lstm_gates_gemm<<<gemmGrid, 256, 0, stream>>>(
            Xbf + (size_t)t * B_ * IH, Hbf, WihB, WhhB, gates);
        lstm_cell<<<(B_ * IH) / 256, 256, 0, stream>>>(
            gates, c0, mask, b_ih, b_hh, cbuf,
            hidden + (size_t)t * B_ * IH, Hbf,
            t, t == 0 ? 1 : 0, t < TT - 1 ? 1 : 0);
    }

    hipMemcpyAsync(h_n, hidden + (size_t)(TT - 1) * B_ * IH,
                   (size_t)B_ * IH * sizeof(float), hipMemcpyDeviceToDevice, stream);
    hipMemcpyAsync(c_n, cbuf,
                   (size_t)B_ * IH * sizeof(float), hipMemcpyDeviceToDevice, stream);
}
